// CrossBandBiMamba_16844861735722
// MI455X (gfx1250) — matrix operand order repaired and checked by reading, not compiled, not run on hardware
//
#include <hip/hip_runtime.h>
#include <hip/hip_bf16.h>
#include <math.h>

// ---------------- problem constants ----------------
#define CHN     128
#define DSTATE  16
#define DCONV   4
#define NBANDS  30
#define DTRANK  8
#define BSZ_    8
#define TT_     500
#define BT_     (BSZ_ * TT_)          // 4000
#define ROWS_   (BT_ * NBANDS)        // 120000

// per-workgroup tile: 2 bt rows * 30 bands = 60 rows, padded to 64 (4 WMMA row tiles)
#define TBT     2
#define VROWS   (TBT * NBANDS)        // 60
#define MROWS   64
#define SBS     136                   // bf16 LDS row stride (272B rows -> 16B-aligned chunks)
#define SXS     48                    // f32 x_dbl LDS row stride (no store guard needed)

// weight element counts
#define IN_W_N   (2 * CHN * CHN)      // 32768
#define X_W_N    ((DTRANK + 2 * DSTATE) * CHN)  // 5120
#define DT_W_N   (CHN * DTRANK)       // 1024
#define OUT_W_N  (CHN * CHN)          // 16384
#define COMB_W_N (CHN * 2 * CHN)      // 32768

typedef __attribute__((ext_vector_type(16))) __bf16 v16bf;
typedef __attribute__((ext_vector_type(8)))  float  v8f;

// ---------------- bf16 helpers (bit-level, compile-safe) ----------------
static __device__ __forceinline__ unsigned short f32_to_bf16(float f) {
    unsigned u = __builtin_bit_cast(unsigned, f);
    unsigned r = u + 0x7FFFu + ((u >> 16) & 1u);
    return (unsigned short)(r >> 16);
}
static __device__ __forceinline__ float bf16_to_f32(unsigned short h) {
    return __builtin_bit_cast(float, ((unsigned)h) << 16);
}
static __device__ __forceinline__ __bf16 us2bf(unsigned short u) {
    return __builtin_bit_cast(__bf16, u);
}

static __device__ __forceinline__ v8f wmma_bf16(v16bf a, v16bf b, v8f c) {
    return __builtin_amdgcn_wmma_f32_16x16x32_bf16(false, a, false, b, (short)0, c, false, false);
}

struct U16x16 { uint4 lo, hi; };   // 32B, reinterpreted as v16bf

// ---------------- WMMA fragment builders (contiguous k = 16*g + s order, wave32) ----------------
// A (16x32 bf16) from LDS: lane m = lane&15, g = lane>>4; two aligned 16B chunks.
static __device__ __forceinline__ v16bf a_frag_lds(const unsigned short* S, int mt,
                                                   int kb, int lane) {
    int m = mt * 16 + (lane & 15);
    int g = lane >> 4;
    const unsigned short* p = S + m * SBS + kb + 16 * g;
    U16x16 u;
    u.lo = *(const uint4*)(p);
    u.hi = *(const uint4*)(p + 8);
    return __builtin_bit_cast(v16bf, u);
}

// A fragment from global bf16 row-major [rows][128] buffer (combine kernel)
static __device__ __forceinline__ v16bf a_frag_glb(const unsigned short* G, int row,
                                                   int kb, int lane) {
    int g = lane >> 4;
    const unsigned short* p = G + (size_t)row * CHN + kb + 16 * g;
    U16x16 u;
    u.lo = *(const uint4*)(p);
    u.hi = *(const uint4*)(p + 8);
    return __builtin_bit_cast(v16bf, u);
}

// B (32x16 bf16) from pre-converted bf16 weight row: 16 contiguous bf16 (32B aligned).
static __device__ __forceinline__ v16bf b_frag_bf(const unsigned short* W, int ldw,
                                                  int n, int kbase) {
    const unsigned short* p = W + n * ldw + kbase;
    U16x16 u;
    u.lo = *(const uint4*)(p);
    u.hi = *(const uint4*)(p + 8);
    return __builtin_bit_cast(v16bf, u);
}

static __device__ __forceinline__ v16bf zero_frag() {
    U16x16 u;
    u.lo = make_uint4(0u, 0u, 0u, 0u);
    u.hi = make_uint4(0u, 0u, 0u, 0u);
    return __builtin_bit_cast(v16bf, u);
}

// B fragment for bf16 dt_w [128][8]: only g==0 & elements 0..7 valid (K=8 zero-padded).
static __device__ __forceinline__ v16bf b_frag_dtw_bf(const unsigned short* W, int n, int g) {
    const unsigned short* p = W + n * DTRANK;          // 16B-aligned row
    uint4 z = make_uint4(0u, 0u, 0u, 0u);
    U16x16 u;
    u.lo = (g == 0) ? *(const uint4*)(p) : z;
    u.hi = z;
    return __builtin_bit_cast(v16bf, u);
}

// A fragment for dt low-rank (f32 LDS cols 0..7): same validity pattern.
static __device__ __forceinline__ v16bf a_frag_dtlr(const float* S, int mt, int lane) {
    int m = mt * 16 + (lane & 15);
    int g = lane >> 4;
    const float* p = S + m * SXS;
    float4 f0 = *(const float4*)(p);
    float4 f1 = *(const float4*)(p + 4);
    float v[8] = {f0.x, f0.y, f0.z, f0.w, f1.x, f1.y, f1.z, f1.w};
    v16bf a = zero_frag();
    if (g == 0) {
#pragma unroll
        for (int i = 0; i < 8; ++i) a[i] = us2bf(f32_to_bf16(v[i]));
    }
    return a;
}

// ---------------- kernel 0: one-time f32 -> bf16 weight conversion ----------------
__global__ __launch_bounds__(256) void cvt_bf16_kernel(const float* __restrict__ s,
                                                       unsigned short* __restrict__ d,
                                                       int n4) {   // n4 = elems/4
    int i = blockIdx.x * 256 + threadIdx.x;
    if (i < n4) {
        float4 f = *(const float4*)(s + i * 4);
        ushort4 o;
        o.x = f32_to_bf16(f.x); o.y = f32_to_bf16(f.y);
        o.z = f32_to_bf16(f.z); o.w = f32_to_bf16(f.w);
        *(ushort4*)(d + i * 4) = o;
    }
}

// ---------------- kernel 1: gather + layernorm -> xn (bf16) ----------------
__global__ __launch_bounds__(256) void prep_ln_kernel(const float* __restrict__ x,
                                                      const float* __restrict__ nw,
                                                      const float* __restrict__ nb,
                                                      unsigned short* __restrict__ xn) {
    int wave = threadIdx.x >> 5, lane = threadIdx.x & 31;
    int r  = blockIdx.x * 8 + wave;          // grid = ROWS_/8 exactly
    int bt = r / NBANDS, k = r - bt * NBANDS;
    int b  = bt / TT_,   t = bt - b * TT_;
    const float* xb = x + (size_t)b * CHN * TT_ * NBANDS + (size_t)t * NBANDS + k;

    float v[4], s = 0.f, s2 = 0.f;
#pragma unroll
    for (int q = 0; q < 4; ++q) {
        int n = lane + 32 * q;
        float f = xb[(size_t)n * (TT_ * NBANDS)];
        v[q] = f; s += f; s2 += f * f;
    }
#pragma unroll
    for (int m = 16; m > 0; m >>= 1) {
        s  += __shfl_xor(s,  m, 32);
        s2 += __shfl_xor(s2, m, 32);
    }
    float mean = s * (1.0f / CHN);
    float var  = s2 * (1.0f / CHN) - mean * mean;
    float rs   = rsqrtf(var + 1e-5f);
    unsigned short* o = xn + (size_t)r * CHN;
#pragma unroll
    for (int q = 0; q < 4; ++q) {
        int n = lane + 32 * q;
        o[n] = f32_to_bf16((v[q] - mean) * rs * nw[n] + nb[n]);
    }
}

// ---------------- kernel 2: fully fused Mamba block, one direction ----------------
template <int REV>
__global__ __launch_bounds__(256) void mamba_dir_kernel(
    const unsigned short* __restrict__ xn, unsigned short* __restrict__ outdir,
    const unsigned short* __restrict__ in_w,  const float* __restrict__ conv_w,
    const float* __restrict__ conv_b, const unsigned short* __restrict__ x_w,
    const unsigned short* __restrict__ dt_w,  const float* __restrict__ dt_b,
    const float* __restrict__ A_log, const float* __restrict__ Dp,
    const unsigned short* __restrict__ out_w) {

    __shared__ unsigned short sA[MROWS * SBS];   // xn tile, later reused as dt
    __shared__ unsigned short sU[MROWS * SBS];   // u (pre- then post-conv)
    __shared__ unsigned short sZ[MROWS * SBS];   // z, later overwritten in place by gated y
    __shared__ float          sXdbl[MROWS * SXS];// x_dbl f32 (dt_lr | B | C | pad)

    int tid = threadIdx.x, wave = tid >> 5, lane = tid & 31;
    int g = lane >> 4;
    int bt0 = blockIdx.x * TBT;

    // ---- load xn tile (vectorized, k reversed for backward direction) ----
    for (int idx = tid; idx < MROWS * 16; idx += 256) {   // 16 uint4-chunks per 128-col row
        int m = idx >> 4, q = idx & 15;
        uint4 v = make_uint4(0u, 0u, 0u, 0u);
        if (m < VROWS) {
            int btl = (m >= NBANDS) ? 1 : 0;
            int k = m - btl * NBANDS;
            int kk = REV ? (NBANDS - 1 - k) : k;
            int gr = (bt0 + btl) * NBANDS + kk;
            v = *(const uint4*)(xn + (size_t)gr * CHN + q * 8);
        }
        *(uint4*)(sA + m * SBS + q * 8) = v;
    }
    __syncthreads();

    // ---- GEMM1: xz = xn @ in_w^T  (64x256, K=128); wave owns 2 col-tiles, B cached ----
    for (int ntl = 0; ntl < 2; ++ntl) {
        int nt = wave * 2 + ntl;                 // 0..15
        int n_out = nt * 16 + (lane & 15);
        v16bf bf[4];
#pragma unroll
        for (int ks = 0; ks < 4; ++ks)
            bf[ks] = b_frag_bf(in_w, CHN, n_out, ks * 32 + 16 * g);
        unsigned short* dst = (nt < 8) ? (sU + n_out) : (sZ + (n_out - CHN));
#pragma unroll
        for (int mt = 0; mt < 4; ++mt) {
            v8f acc = {};
#pragma unroll
            for (int ks = 0; ks < 4; ++ks)
                acc = wmma_bf16(a_frag_lds(sA, mt, ks * 32, lane), bf[ks], acc);
#pragma unroll
            for (int j = 0; j < 8; ++j)
                dst[(mt * 16 + j + 8 * g) * SBS] = f32_to_bf16(acc[j]);
        }
    }
    __syncthreads();

    // ---- causal depthwise conv (DC=4) + SiLU, in registers, written back in place ----
    {
        int btl = tid >> 7, n = tid & 127;
        float w0 = conv_w[n * DCONV + 0], w1 = conv_w[n * DCONV + 1];
        float w2 = conv_w[n * DCONV + 2], w3 = conv_w[n * DCONV + 3];
        float cb = conv_b[n];
        float vals[NBANDS];
        int base = btl * NBANDS;
#pragma unroll
        for (int k = 0; k < NBANDS; ++k) {
            float acc = cb + w3 * bf16_to_f32(sU[(base + k) * SBS + n]);
            if (k >= 1) acc += w2 * bf16_to_f32(sU[(base + k - 1) * SBS + n]);
            if (k >= 2) acc += w1 * bf16_to_f32(sU[(base + k - 2) * SBS + n]);
            if (k >= 3) acc += w0 * bf16_to_f32(sU[(base + k - 3) * SBS + n]);
            vals[k] = acc / (1.0f + expf(-acc));       // silu
        }
        __syncthreads();
#pragma unroll
        for (int k = 0; k < NBANDS; ++k)
            sU[(base + k) * SBS + n] = f32_to_bf16(vals[k]);
    }
    __syncthreads();

    // ---- GEMM2: x_dbl = u @ x_w^T  (64x40->48, K=128); waves 0..2, B cached+clamped ----
    if (wave < 3) {
        int nt = wave;
        int n = nt * 16 + (lane & 15);           // 0..47
        int nn = (n < DTRANK + 2 * DSTATE) ? n : 0;   // clamp row -> unguarded loads
        v16bf bf[4];
#pragma unroll
        for (int ks = 0; ks < 4; ++ks) {
            v16bf b = b_frag_bf(x_w, CHN, nn, ks * 32 + 16 * g);
            bf[ks] = (n < DTRANK + 2 * DSTATE) ? b : zero_frag();
        }
#pragma unroll
        for (int mt = 0; mt < 4; ++mt) {
            v8f acc = {};
#pragma unroll
            for (int ks = 0; ks < 4; ++ks)
                acc = wmma_bf16(a_frag_lds(sU, mt, ks * 32, lane), bf[ks], acc);
#pragma unroll
            for (int j = 0; j < 8; ++j)
                sXdbl[(mt * 16 + j + 8 * g) * SXS + n] = acc[j];
        }
    }
    __syncthreads();

    // ---- GEMM3: dt = softplus(dt_lr @ dt_w^T + dt_b) (64x128, K=8 zero-padded) ----
    unsigned short* sDT = sA;   // xn tile no longer needed
    {
        int nt = wave;                           // 8 waves = 8 col tiles
        int n = nt * 16 + (lane & 15);
        v16bf b = b_frag_dtw_bf(dt_w, n, g);
        float db = dt_b[n];
#pragma unroll
        for (int mt = 0; mt < 4; ++mt) {
            v8f acc = {};
            acc = wmma_bf16(a_frag_dtlr(sXdbl, mt, lane), b, acc);
#pragma unroll
            for (int j = 0; j < 8; ++j) {
                float v = acc[j] + db;
                float sp = (v > 20.0f) ? v : log1pf(expf(v));
                sDT[(mt * 16 + j + 8 * g) * SBS + n] = f32_to_bf16(sp);
            }
        }
    }
    __syncthreads();

    // ---- selective scan over L=30, 16 states per thread, one (bt,n) pair per thread ----
    {
        int btl = tid >> 7, n = tid & 127;
        float negA[DSTATE], h[DSTATE];
#pragma unroll
        for (int s = 0; s < DSTATE; ++s) {
            negA[s] = -expf(A_log[n * DSTATE + s]);
            h[s] = 0.0f;
        }
        float Dn = Dp[n];
        for (int k = 0; k < NBANDS; ++k) {
            int row = btl * NBANDS + k;
            float dt = bf16_to_f32(sDT[row * SBS + n]);
            float u  = bf16_to_f32(sU[row * SBS + n]);
            float z  = bf16_to_f32(sZ[row * SBS + n]);
            float dtu = dt * u;
            float y = 0.0f;
            const float* xd = sXdbl + row * SXS;
#pragma unroll
            for (int s = 0; s < DSTATE; ++s) {
                float dA = __expf(dt * negA[s]);
                h[s] = dA * h[s] + dtu * xd[DTRANK + s];           // B_t
                y += h[s] * xd[DTRANK + DSTATE + s];               // C_t
            }
            y += u * Dn;
            y *= z / (1.0f + expf(-z));                            // y * silu(z)
            sZ[row * SBS + n] = f32_to_bf16(y);                    // in-place: z -> gated y
        }
    }
    __syncthreads();

    // ---- GEMM4: out = y @ out_w^T (64x128, K=128); wave owns 1 col-tile, B cached ----
    {
        int nt = wave;
        int n = nt * 16 + (lane & 15);
        v16bf bf[4];
#pragma unroll
        for (int ks = 0; ks < 4; ++ks)
            bf[ks] = b_frag_bf(out_w, CHN, n, ks * 32 + 16 * g);
#pragma unroll
        for (int mt = 0; mt < 4; ++mt) {
            v8f acc = {};
#pragma unroll
            for (int ks = 0; ks < 4; ++ks)
                acc = wmma_bf16(a_frag_lds(sZ, mt, ks * 32, lane), bf[ks], acc);
            int m0 = mt * 16 + 8 * g;
#pragma unroll
            for (int j = 0; j < 8; ++j) {
                int m = m0 + j;
                if (m < VROWS) {
                    int btl = (m >= NBANDS) ? 1 : 0;       // m < 60: compare, not divide
                    int k = m - btl * NBANDS;
                    int kk = REV ? (NBANDS - 1 - k) : k;
                    int gr = (bt0 + btl) * NBANDS + kk;
                    outdir[(size_t)gr * CHN + n] = f32_to_bf16(acc[j]);
                }
            }
        }
    }
}

// ---------------- kernel 3: combine GEMM (K=256) + bias + residual + scatter ----------------
__global__ __launch_bounds__(256) void combine_kernel(
    const unsigned short* __restrict__ outf, const unsigned short* __restrict__ outb,
    const unsigned short* __restrict__ comb_w, const float* __restrict__ comb_b,
    const float* __restrict__ x, float* __restrict__ out) {

    int tid = threadIdx.x, wave = tid >> 5, lane = tid & 31;
    int g = lane >> 4;
    int row0 = blockIdx.x * 64;                       // grid = ROWS_/64 exactly

    int nt = wave;                                    // 8 waves = 8 col tiles
    int n = nt * 16 + (lane & 15);
    v8f acc[4] = {{}, {}, {}, {}};

    for (int ks = 0; ks < 8; ++ks) {                  // K = 256: 4 from fwd, 4 from bwd
        v16bf b = b_frag_bf(comb_w, 2 * CHN, n, ks * 32 + 16 * g);
        const unsigned short* src = (ks < 4) ? outf : outb;
        int kb = (ks & 3) * 32;
#pragma unroll
        for (int mt = 0; mt < 4; ++mt) {
            v16bf a = a_frag_glb(src, row0 + mt * 16 + (lane & 15), kb, lane);
            acc[mt] = wmma_bf16(a, b, acc[mt]);
        }
    }

    float bias = comb_b[n];
#pragma unroll
    for (int mt = 0; mt < 4; ++mt) {
        // two divides per row-tile; then branchless incremental (k, t, b, idx)
        int gr = row0 + mt * 16 + 8 * g;              // 8 consecutive rows from here
        int bt = gr / NBANDS,  k  = gr - bt * NBANDS;
        int b_ = bt / TT_,     tt = bt - b_ * TT_;
        size_t idx = (((size_t)(b_ * CHN + n)) * TT_ + tt) * NBANDS + k;
#pragma unroll
        for (int j = 0; j < 8; ++j) {
            out[idx] = acc[mt][j] + bias + x[idx];
            // advance one row: (t,k) flatten contiguously; fix-up only when t wraps
            ++k; ++idx;
            bool wk = (k == NBANDS);
            k = wk ? 0 : k;
            tt += wk ? 1 : 0;
            bool wt = (tt == TT_);
            tt = wt ? 0 : tt;
            idx += wt ? (size_t)(CHN - 1) * TT_ * NBANDS : 0;
        }
    }
}

// ---------------- host launcher ----------------
extern "C" void kernel_launch(void* const* d_in, const int* in_sizes, int n_in,
                              void* d_out, int out_size, void* d_ws, size_t ws_size,
                              hipStream_t stream) {
    (void)in_sizes; (void)n_in; (void)out_size; (void)ws_size;

    const float* x      = (const float*)d_in[0];
    const float* norm_w = (const float*)d_in[1];
    const float* norm_b = (const float*)d_in[2];
    // fwd params: d_in[3..11], bwd params: d_in[12..20]
    const float* comb_w = (const float*)d_in[21];
    const float* comb_b = (const float*)d_in[22];

    unsigned short* ws   = (unsigned short*)d_ws;
    const size_t NPER    = (size_t)ROWS_ * CHN;       // bf16 elements per activation buffer
    unsigned short* xnbf = ws;
    unsigned short* outf = ws + NPER;
    unsigned short* outb = ws + 2 * NPER;
    unsigned short* wp   = ws + 3 * NPER;             // bf16 weight area
    unsigned short* wf_in  = wp;               wp += IN_W_N;
    unsigned short* wf_x   = wp;               wp += X_W_N;
    unsigned short* wf_dt  = wp;               wp += DT_W_N;
    unsigned short* wf_out = wp;               wp += OUT_W_N;
    unsigned short* wb_in  = wp;               wp += IN_W_N;
    unsigned short* wb_x   = wp;               wp += X_W_N;
    unsigned short* wb_dt  = wp;               wp += DT_W_N;
    unsigned short* wb_out = wp;               wp += OUT_W_N;
    unsigned short* w_cmb  = wp;               wp += COMB_W_N;

    // one-time weight conversion (deterministic, replayed each call; trivially cheap)
    auto cvt = [&](const float* s, unsigned short* d, int nelem) {
        int n4 = nelem / 4;
        cvt_bf16_kernel<<<(n4 + 255) / 256, 256, 0, stream>>>(s, d, n4);
    };
    cvt((const float*)d_in[3],  wf_in,  IN_W_N);
    cvt((const float*)d_in[6],  wf_x,   X_W_N);
    cvt((const float*)d_in[7],  wf_dt,  DT_W_N);
    cvt((const float*)d_in[11], wf_out, OUT_W_N);
    cvt((const float*)d_in[12], wb_in,  IN_W_N);
    cvt((const float*)d_in[15], wb_x,   X_W_N);
    cvt((const float*)d_in[16], wb_dt,  DT_W_N);
    cvt((const float*)d_in[20], wb_out, OUT_W_N);
    cvt(comb_w,                 w_cmb,  COMB_W_N);

    prep_ln_kernel<<<ROWS_ / 8, 256, 0, stream>>>(x, norm_w, norm_b, xnbf);

    mamba_dir_kernel<0><<<BT_ / TBT, 256, 0, stream>>>(
        xnbf, outf,
        wf_in, (const float*)d_in[4], (const float*)d_in[5],
        wf_x, wf_dt, (const float*)d_in[8],
        (const float*)d_in[9], (const float*)d_in[10], wf_out);

    mamba_dir_kernel<1><<<BT_ / TBT, 256, 0, stream>>>(
        xnbf, outb,
        wb_in, (const float*)d_in[13], (const float*)d_in[14],
        wb_x, wb_dt, (const float*)d_in[17],
        (const float*)d_in[18], (const float*)d_in[19], wb_out);

    combine_kernel<<<ROWS_ / 64, 256, 0, stream>>>(outf, outb, w_cmb, comb_b,
                                                   x, (float*)d_out);
}
